// Q_network_51350628991319
// MI455X (gfx1250) — matrix operand order repaired and checked by reading, not compiled, not run on hardware
//
#include <hip/hip_runtime.h>
#include <math.h>

typedef __attribute__((ext_vector_type(16))) _Float16 v16h;
typedef __attribute__((ext_vector_type(8)))  float    v8f;

#define WMMA32(a,b,c) __builtin_amdgcn_wmma_f32_16x16x32_f16(false,(a),false,(b),(short)0,(c),false,false)

// ---------------- problem constants ----------------
#define B_  8
#define T_  100
#define N_  20
#define E_  20
#define BT_ 800
#define ALLYD 128
#define ENEMYD 128
#define A_  12
#define H_  4
#define DH_ 16
#define D_  64
#define M_  19
#define NROW 16000          // BT*N
#define BN_ 160             // B*N
#define FFNH 2048

// ---------------- workspace layout (float offsets) ----------------
#define OFF_ST   ((size_t)0)            // 16000*128
#define OFF_SE   ((size_t)2048000)      // 16000*64
#define OFF_OAE  ((size_t)3072000)      // 16000*64
#define OFF_KP   ((size_t)4096000)      // 800*4*20*64
#define OFF_QP   ((size_t)8192000)      // 800*4*20*64
#define OFF_LQ   ((size_t)12288000)     // 800*20*2
#define OFF_LK   ((size_t)12320000)     // 800*20*2
#define OFF_WSF  ((size_t)12352000)     // 800*4*20*19 (w_soft)
#define OFF_VP   ((size_t)13568000)     // 800*4*20*16
#define OFF_AGG  ((size_t)14592000)     // 16000*64
#define OFF_X1   ((size_t)15616000)     // 16000*64
#define OFF_X2   ((size_t)16640000)     // 16000*64
#define OFF_EE   ((size_t)17664000)     // 800*64
#define OFF_GX   ((size_t)17715200)     // 100*160*192
#define OFF_HST  ((size_t)20787200)     // 160*64
#define OFF_YS   ((size_t)20797440)     // 100*160*64
#define OFF_F16  ((size_t)21821440)     // f16 weights (as float slots)

// ---------------- output layout (float offsets into d_out) ----------------
#define OQV ((size_t)0)                 // 16000
#define OW  ((size_t)16000)             // 800*4*20*20
#define OS  ((size_t)1296000)           // 800*4*20*19
#define OH  ((size_t)2512000)           // 160*64

// ---------------- helpers ----------------
__device__ __forceinline__ float gelu_f(float x){
  return 0.5f * x * (1.0f + erff(x * 0.70710678118654752f));
}
__device__ __forceinline__ float sigm_f(float x){ return 1.0f/(1.0f+expf(-x)); }

// blockDim.x must equal count (power of two); all threads call.
__device__ __forceinline__ float block_sum(float v, float* red, int count){
  int tid = threadIdx.x;
  red[tid] = v; __syncthreads();
  for (int s = count >> 1; s > 0; s >>= 1){
    if (tid < s) red[tid] += red[tid + s];
    __syncthreads();
  }
  float r = red[0]; __syncthreads();
  return r;
}

__device__ __forceinline__ float hash_u(unsigned int x){
  x ^= 61u ^ (x >> 16); x *= 9u; x ^= x >> 4; x *= 0x27d4eb2du; x ^= x >> 15;
  return ((float)((x >> 8) & 0xFFFFFFu) + 0.5f) * (1.0f/16777216.0f);
}
__device__ __forceinline__ float gumbel_f(unsigned int i){
  float u = hash_u(i * 0x9E3779B9u + 0x85ebca6bu);
  return -logf(-logf(u));
}

// ---- WMMA fragment loaders (ISA 7.12.2 layouts, wave32) ----
// A: 16x32 f16 tile from LDS (row-major, leading dim ld)
__device__ __forceinline__ v16h ldsA_frag(const _Float16* src, int ld, int k0, int lane){
  v16h a;
  int m  = lane & 15;
  int kh = (lane >> 4) * 8;
  #pragma unroll
  for (int g = 0; g < 8; ++g){
    int base = k0 + ((g & 3) * 2) + ((g >> 2) * 16) + kh;
    a[2*g]   = src[m*ld + base];
    a[2*g+1] = src[m*ld + base + 1];
  }
  return a;
}
// B: 32x16 f16 tile where elem(k,n) = w[(n0+n)*ldw + k]   (i.e. B = W^T, W row-major)
__device__ __forceinline__ v16h globB_frag(const _Float16* w, int ldw, int n0, int k0, int lane){
  v16h b;
  int n  = n0 + (lane & 15);
  int kh = (lane >> 4) * 8;
  const _Float16* r = w + (size_t)n * ldw;
  #pragma unroll
  for (int g = 0; g < 8; ++g){
    int k = k0 + kh + ((g & 3) * 2) + ((g >> 2) * 16);
    b[2*g]   = r[k];
    b[2*g+1] = r[k + 1];
  }
  return b;
}

// ================= kernels =================

// convert FFN weights to f16
__global__ void k_cvt(const float* __restrict__ w1, const float* __restrict__ w2,
                      _Float16* __restrict__ w1h, _Float16* __restrict__ w2h){
  int i = blockIdx.x * blockDim.x + threadIdx.x;
  if (i < FFNH*D_){ w1h[i] = (_Float16)w1[i]; w2h[i] = (_Float16)w2[i]; }
}

// st = ln(states); se = gelu(ln(st @ w_ally1^T + b))
__global__ __launch_bounds__(128) void k_st_se(const float* __restrict__ states,
    const float* __restrict__ w, const float* __restrict__ b,
    float* __restrict__ st, float* __restrict__ se){
  __shared__ float sbuf[ALLYD]; __shared__ float red[128];
  int row = blockIdx.x, tid = threadIdx.x;
  float x  = states[(size_t)row*ALLYD + tid];
  float mu = block_sum(x, red, 128) * (1.0f/128.0f);
  float d  = x - mu;
  float var= block_sum(d*d, red, 128) * (1.0f/128.0f);
  float sv = d * rsqrtf(var + 1e-5f);
  st[(size_t)row*ALLYD + tid] = sv;
  sbuf[tid] = sv; __syncthreads();
  float y = 0.f;
  if (tid < D_){
    const float* wr = w + tid*ALLYD;
    for (int k = 0; k < ALLYD; ++k) y += sbuf[k]*wr[k];
    y += b[tid];
  }
  float mu2  = block_sum(tid < D_ ? y : 0.f, red, 128) * (1.0f/64.0f);
  float d2   = y - mu2;
  float var2 = block_sum(tid < D_ ? d2*d2 : 0.f, red, 128) * (1.0f/64.0f);
  if (tid < D_) se[(size_t)row*D_ + tid] = gelu_f(d2 * rsqrtf(var2 + 1e-5f));
}

// oae = gelu(ln(cat(st,act) @ w_allyact^T + b))
__global__ __launch_bounds__(128) void k_oae(const float* __restrict__ st,
    const float* __restrict__ actions, const float* __restrict__ w, const float* __restrict__ b,
    float* __restrict__ oae){
  __shared__ float cbuf[ALLYD + A_]; __shared__ float red[128];
  int row = blockIdx.x, tid = threadIdx.x;
  cbuf[tid] = st[(size_t)row*ALLYD + tid];
  if (tid < A_) cbuf[ALLYD + tid] = actions[(size_t)row*A_ + tid];
  __syncthreads();
  float y = 0.f;
  if (tid < D_){
    const float* wr = w + tid*(ALLYD + A_);
    for (int k = 0; k < ALLYD + A_; ++k) y += cbuf[k]*wr[k];
    y += b[tid];
  }
  float mu  = block_sum(tid < D_ ? y : 0.f, red, 128) * (1.0f/64.0f);
  float d   = y - mu;
  float var = block_sum(tid < D_ ? d*d : 0.f, red, 128) * (1.0f/64.0f);
  if (tid < D_) oae[(size_t)row*D_ + tid] = gelu_f(d * rsqrtf(var + 1e-5f));
}

// ee = gelu(ln(ln(enemy).flatten @ w_enemy^T + b)), one block per bt
__global__ __launch_bounds__(128) void k_ee(const float* __restrict__ enemy,
    const float* __restrict__ w, const float* __restrict__ b, float* __restrict__ ee){
  __shared__ float buf[E_*ENEMYD]; __shared__ float red[128];
  int bt = blockIdx.x, tid = threadIdx.x;
  for (int e = 0; e < E_; ++e){
    float x  = enemy[((size_t)bt*E_ + e)*ENEMYD + tid];
    float mu = block_sum(x, red, 128) * (1.0f/128.0f);
    float d  = x - mu;
    float var= block_sum(d*d, red, 128) * (1.0f/128.0f);
    buf[e*ENEMYD + tid] = d * rsqrtf(var + 1e-5f);
  }
  __syncthreads();
  float y = 0.f;
  if (tid < D_){
    const float* wr = w + (size_t)tid*(E_*ENEMYD);
    for (int k = 0; k < E_*ENEMYD; ++k) y += buf[k]*wr[k];
    y += b[tid];
  }
  float mu  = block_sum(tid < D_ ? y : 0.f, red, 128) * (1.0f/64.0f);
  float d   = y - mu;
  float var = block_sum(tid < D_ ? d*d : 0.f, red, 128) * (1.0f/64.0f);
  if (tid < D_) ee[(size_t)bt*D_ + tid] = gelu_f(d * rsqrtf(var + 1e-5f));
}

// per-(bt,agent) projections: K,Q,V + hard-attn partial logits Lq,Lk
__global__ __launch_bounds__(64) void k_proj(const float* __restrict__ se, const float* __restrict__ oae,
    const float* __restrict__ kw, const float* __restrict__ kb,
    const float* __restrict__ qw, const float* __restrict__ qb,
    const float* __restrict__ avw, const float* __restrict__ avb,
    const float* __restrict__ hw, const float* __restrict__ hlw,
    float* __restrict__ Kp, float* __restrict__ Qp, float* __restrict__ Vp,
    float* __restrict__ Lq, float* __restrict__ Lk){
  __shared__ float sse[D_], soa[D_];
  __shared__ float sK[H_][D_], sQ[H_][D_], sHq[H_][DH_], sHk[H_][DH_];
  int row = blockIdx.x, tid = threadIdx.x;
  int bt = row / N_, j = row % N_;
  sse[tid] = se[(size_t)row*D_ + tid];
  soa[tid] = oae[(size_t)row*D_ + tid];
  __syncthreads();
  for (int h = 0; h < H_; ++h){
    const float* kr = kw + ((size_t)(h*D_ + tid))*D_;
    const float* qr = qw + ((size_t)(h*D_ + tid))*D_;
    float kv = kb[h*D_ + tid], qv = qb[h*D_ + tid];
    for (int c = 0; c < D_; ++c){ kv += sse[c]*kr[c]; qv += sse[c]*qr[c]; }
    size_t idx = (((size_t)bt*H_ + h)*N_ + j)*D_ + tid;
    Kp[idx] = kv; Qp[idx] = qv;
    sK[h][tid] = kv; sQ[h][tid] = qv;
  }
  __syncthreads();
  for (int h = 0; h < H_; ++h){
    if (tid < DH_){
      const float* hr = hw + ((size_t)(h*DH_ + tid))*(2*D_);
      float hq = 0.f, hk = 0.f;
      for (int c = 0; c < D_; ++c){ hq += sQ[h][c]*hr[c]; hk += sK[h][c]*hr[D_ + c]; }
      sHq[h][tid] = hq; sHk[h][tid] = hk;
      const float* ar = avw + ((size_t)(h*DH_ + tid))*D_;
      float vv = avb[h*DH_ + tid];
      for (int c = 0; c < D_; ++c) vv += soa[c]*ar[c];
      Vp[(((size_t)bt*H_ + h)*N_ + j)*DH_ + tid] = vv;
    }
  }
  __syncthreads();
  if (tid < 2){
    float lq = 0.f, lk = 0.f;
    for (int i = 0; i < H_*DH_; ++i){
      float wv = hlw[tid*(H_*DH_) + i];
      lq += sHq[i >> 4][i & 15] * wv;
      lk += sHk[i >> 4][i & 15] * wv;
    }
    Lq[(size_t)row*2 + tid] = lq;
    Lk[(size_t)row*2 + tid] = lk;
  }
}

// per-(bt,h,n): score, gumbel hard mask, softmax, weights out, agg
__global__ __launch_bounds__(32) void k_attn(const float* __restrict__ Kp, const float* __restrict__ Qp,
    const float* __restrict__ Vp, const float* __restrict__ Lq, const float* __restrict__ Lk,
    const float* __restrict__ hb, const float* __restrict__ hlw, const float* __restrict__ hlb,
    float* __restrict__ wsoft, float* __restrict__ aggb,
    float* __restrict__ out_w, float* __restrict__ out_s){
  __shared__ float ssc[32], sws[32], sLb[2];
  int id = blockIdx.x, tid = threadIdx.x;
  int n  = id % N_; int h = (id / N_) % H_; int bt = id / (N_*H_);
  if (tid < 2){
    float s = hlb[tid];
    for (int i = 0; i < H_*DH_; ++i) s += hb[i]*hlw[tid*(H_*DH_) + i];
    sLb[tid] = s;
  }
  int m = tid; bool act = (m < M_);
  int j = (m < n) ? m : m + 1;
  float sc = 0.f;
  if (act){
    const float* q = Qp + (((size_t)bt*H_ + h)*N_ + n)*D_;
    const float* k = Kp + (((size_t)bt*H_ + h)*N_ + j)*D_;
    for (int e = 0; e < D_; ++e) sc += q[e]*k[e];
    sc *= 0.125f; // 1/sqrt(64)
    out_s[(((size_t)bt*H_ + h)*N_ + n)*M_ + m] = sc;
  }
  __syncthreads();
  float haw = 0.f;
  if (act){
    unsigned int base = (((unsigned int)(bt*N_ + n))*M_ + (unsigned int)m)*2u;
    float l0 = Lq[((size_t)bt*N_ + n)*2 + 0] + Lk[((size_t)bt*N_ + j)*2 + 0] + sLb[0] + gumbel_f(base);
    float l1 = Lq[((size_t)bt*N_ + n)*2 + 1] + Lk[((size_t)bt*N_ + j)*2 + 1] + sLb[1] + gumbel_f(base + 1u);
    haw = (l1 > l0) ? 1.f : 0.f;
  }
  ssc[tid] = act ? sc : -3.0e38f; __syncthreads();
  float mx = -3.0e38f;
  for (int t = 0; t < M_; ++t) mx = fmaxf(mx, ssc[t]);
  float ex = act ? expf(sc - mx) : 0.f;
  __syncthreads(); sws[tid] = ex; __syncthreads();
  float sum = 0.f;
  for (int t = 0; t < M_; ++t) sum += sws[t];
  float wv = act ? (ex / sum) * haw : 0.f;
  __syncthreads(); sws[tid] = wv; __syncthreads();
  if (act) wsoft[(((size_t)bt*H_ + h)*N_ + n)*M_ + m] = wv;
  if (tid < N_){
    float val = (tid == n) ? 1.f : sws[(tid < n) ? tid : tid - 1];
    out_w[(((size_t)bt*H_ + h)*N_ + n)*N_ + tid] = val;
  }
  if (tid < DH_){
    float a = 0.f;
    for (int m2 = 0; m2 < M_; ++m2){
      int j2 = (m2 < n) ? m2 : m2 + 1;
      a += sws[m2] * Vp[(((size_t)bt*H_ + h)*N_ + j2)*DH_ + tid];
    }
    aggb[((size_t)bt*N_ + n)*D_ + h*DH_ + tid] = a;
  }
}

// x1 = ln(oae + agg)
__global__ __launch_bounds__(64) void k_x1(const float* __restrict__ oae, const float* __restrict__ aggb,
                                           float* __restrict__ x1){
  __shared__ float red[64];
  int row = blockIdx.x, tid = threadIdx.x;
  float v   = oae[(size_t)row*D_ + tid] + aggb[(size_t)row*D_ + tid];
  float mu  = block_sum(v, red, 64) * (1.0f/64.0f);
  float d   = v - mu;
  float var = block_sum(d*d, red, 64) * (1.0f/64.0f);
  x1[(size_t)row*D_ + tid] = d * rsqrtf(var + 1e-5f);
}

// FFN: x2 = ln(x1 + gelu(ln(x1 W1^T + b1)) W2^T + b2)   — WMMA f16, f32 accum.
// Async global->LDS copy of the x1 tile; GEMM1 accumulators retained in VGPRs
// across the LN-statistics pass so phase 2 never re-reads W1.
__global__ __launch_bounds__(256) void k_ffn(const float* __restrict__ x1,
    const _Float16* __restrict__ w1h, const float* __restrict__ fb1,
    const _Float16* __restrict__ w2h, const float* __restrict__ fb2,
    float* __restrict__ x2){
  __shared__ float    sX[16*64];
  __shared__ _Float16 sA[16*64];
  __shared__ float    sStats[32];
  __shared__ float    sY[16*64];
  __shared__ _Float16 sStg[8][16*32];
  const int tid = threadIdx.x, wave = tid >> 5, lane = tid & 31;
  const int row0 = blockIdx.x * 16;

  // ---- async copy of the 16x64 f32 x1 tile (4 KB): one b128 per thread ----
  {
    unsigned lds_off = (unsigned)(unsigned long long)(&sX[0]) + (unsigned)tid * 16u;
    unsigned long long gaddr =
        (unsigned long long)(const void*)(x1 + (size_t)row0*64 + (size_t)tid*4);
    asm volatile("global_load_async_to_lds_b128 %0, %1, off"
                 :: "v"(lds_off), "v"(gaddr) : "memory");
  }
  if (tid < 32) sStats[tid] = 0.f;
  __builtin_prefetch(w1h + (size_t)wave*256*64, 0, 1);
  __builtin_prefetch(w2h + (size_t)wave*256, 0, 1);
  asm volatile("s_wait_asynccnt 0x0" ::: "memory");
  __syncthreads();
  for (int i = tid; i < 16*64; i += 256){ sA[i] = (_Float16)sX[i]; sY[i] = 0.f; }
  __syncthreads();

  v16h a0 = ldsA_frag(sA, 64, 0,  lane);
  v16h a1 = ldsA_frag(sA, 64, 32, lane);

  // ---- phase 1: GEMM1 over wave's 256-col slice; keep accs, gather LN stats ----
  v8f acc[16];
  float rs[8], rq[8];
  #pragma unroll
  for (int g = 0; g < 8; ++g){ rs[g] = 0.f; rq[g] = 0.f; }
  #pragma unroll
  for (int nt = 0; nt < 16; ++nt){
    int col0 = wave*256 + nt*16;
    v8f a_ = {};
    a_ = WMMA32(a0, globB_frag(w1h, 64, col0, 0,  lane), a_);
    a_ = WMMA32(a1, globB_frag(w1h, 64, col0, 32, lane), a_);
    float bias = fb1[col0 + (lane & 15)];
    #pragma unroll
    for (int g = 0; g < 8; ++g){
      float v = a_[g] + bias; a_[g] = v;
      rs[g] += v; rq[g] += v*v;
    }
    acc[nt] = a_;
  }
  #pragma unroll
  for (int g = 0; g < 8; ++g){
    int r = g + 8*(lane >> 4);
    atomicAdd(&sStats[r*2 + 0], rs[g]);
    atomicAdd(&sStats[r*2 + 1], rq[g]);
  }
  __syncthreads();
  if (tid < 16){
    float mu  = sStats[tid*2]     * (1.0f/2048.0f);
    float var = sStats[tid*2 + 1] * (1.0f/2048.0f) - mu*mu;
    sStats[tid*2] = mu; sStats[tid*2 + 1] = rsqrtf(var + 1e-5f);
  }
  __syncthreads();

  // ---- phase 2: LN+GELU the retained accs, feed GEMM2 k-slices ----
  v8f y0 = {}, y1 = {}, y2 = {}, y3 = {};
  _Float16* stg = sStg[wave];
  #pragma unroll
  for (int cb = 0; cb < 8; ++cb){
    int kg0 = wave*256 + cb*32;
    #pragma unroll
    for (int hh = 0; hh < 2; ++hh){
      v8f a_ = acc[cb*2 + hh];
      #pragma unroll
      for (int g = 0; g < 8; ++g){
        int r = g + 8*(lane >> 4);
        float v = (a_[g] - sStats[r*2]) * sStats[r*2 + 1];
        stg[r*32 + hh*16 + (lane & 15)] = (_Float16)gelu_f(v);
      }
    }
    __syncthreads();
    v16h ah = ldsA_frag(stg, 32, 0, lane);
    y0 = WMMA32(ah, globB_frag(w2h, 2048, 0,  kg0, lane), y0);
    y1 = WMMA32(ah, globB_frag(w2h, 2048, 16, kg0, lane), y1);
    y2 = WMMA32(ah, globB_frag(w2h, 2048, 32, kg0, lane), y2);
    y3 = WMMA32(ah, globB_frag(w2h, 2048, 48, kg0, lane), y3);
    __syncthreads();
  }
  #pragma unroll
  for (int g = 0; g < 8; ++g){
    int r = g + 8*(lane >> 4), c = lane & 15;
    atomicAdd(&sY[r*64 + 0  + c], y0[g]);
    atomicAdd(&sY[r*64 + 16 + c], y1[g]);
    atomicAdd(&sY[r*64 + 32 + c], y2[g]);
    atomicAdd(&sY[r*64 + 48 + c], y3[g]);
  }
  __syncthreads();
  // ---- epilogue: x2 = ln(x1 + y + b2) ----
  if (tid < 16){
    float t[64]; float s = 0.f, q = 0.f;
    for (int c = 0; c < 64; ++c){ float v = sX[tid*64 + c] + sY[tid*64 + c] + fb2[c]; t[c] = v; s += v; }
    float mu = s * (1.0f/64.0f);
    for (int c = 0; c < 64; ++c){ float d = t[c] - mu; q += d*d; }
    float rstd = rsqrtf(q * (1.0f/64.0f) + 1e-5f);
    for (int c = 0; c < 64; ++c) x2[((size_t)row0 + tid)*64 + c] = (t[c] - mu)*rstd;
  }
}

// common1/common2 + GRU input precompute gx = c2 @ wih^T + bih
__global__ __launch_bounds__(128) void k_common(const float* __restrict__ se, const float* __restrict__ ee,
    const float* __restrict__ x2, const float* __restrict__ c1w, const float* __restrict__ c1b,
    const float* __restrict__ c2w, const float* __restrict__ c2b,
    const float* __restrict__ gwih, const float* __restrict__ gbih, float* __restrict__ gxall){
  __shared__ float cat[3*D_]; __shared__ float c1[128]; __shared__ float red[128]; __shared__ float c2s[D_];
  int row = blockIdx.x, tid = threadIdx.x;
  int bt = row / N_, n = row % N_;
  if (tid < D_){
    cat[tid]        = se[(size_t)row*D_ + tid];
    cat[D_ + tid]   = ee[(size_t)bt*D_ + tid];
    cat[2*D_ + tid] = x2[(size_t)row*D_ + tid];
  }
  __syncthreads();
  float y = c1b[tid];
  { const float* wr = c1w + (size_t)tid*(3*D_);
    for (int k = 0; k < 3*D_; ++k) y += cat[k]*wr[k]; }
  float mu  = block_sum(y, red, 128) * (1.0f/128.0f);
  float d   = y - mu;
  float var = block_sum(d*d, red, 128) * (1.0f/128.0f);
  c1[tid] = gelu_f(d * rsqrtf(var + 1e-5f));
  __syncthreads();
  float y2v = 0.f;
  if (tid < D_){
    const float* wr = c2w + (size_t)tid*128;
    y2v = c2b[tid];
    for (int k = 0; k < 128; ++k) y2v += c1[k]*wr[k];
  }
  float mu2  = block_sum(tid < D_ ? y2v : 0.f, red, 128) * (1.0f/64.0f);
  float d2   = y2v - mu2;
  float var2 = block_sum(tid < D_ ? d2*d2 : 0.f, red, 128) * (1.0f/64.0f);
  if (tid < D_) c2s[tid] = gelu_f(d2 * rsqrtf(var2 + 1e-5f));
  __syncthreads();
  int b = bt / T_, t = bt % T_;
  int bn = b*N_ + n;
  float* gx = gxall + ((size_t)t*BN_ + bn)*(3*D_);
  for (int g = tid; g < 3*D_; g += 128){
    float s = gbih[g];
    const float* wr = gwih + (size_t)g*D_;
    for (int k = 0; k < D_; ++k) s += c2s[k]*wr[k];
    gx[g] = s;
  }
}

// one GRU step (prev may be the rnn input for t==0, or hstate)
__global__ __launch_bounds__(64) void k_gru(const float* __restrict__ prev, const float* __restrict__ gxall,
    const float* __restrict__ whh, const float* __restrict__ bhh,
    float* __restrict__ hstate, float* __restrict__ ys, int t){
  __shared__ float h[D_]; __shared__ float gh[3*D_];
  int bn = blockIdx.x, tid = threadIdx.x;
  h[tid] = prev[(size_t)bn*D_ + tid];
  __syncthreads();
  for (int g = tid; g < 3*D_; g += 64){
    float s = bhh[g];
    const float* wr = whh + (size_t)g*D_;
    for (int k = 0; k < D_; ++k) s += h[k]*wr[k];
    gh[g] = s;
  }
  __syncthreads();
  const float* gx = gxall + ((size_t)t*BN_ + bn)*(3*D_);
  float r  = sigm_f(gx[tid]        + gh[tid]);
  float z  = sigm_f(gx[D_ + tid]   + gh[D_ + tid]);
  float nn = tanhf (gx[2*D_ + tid] + r*gh[2*D_ + tid]);
  float hn = (1.f - z)*nn + z*h[tid];
  hstate[(size_t)bn*D_ + tid] = hn;
  ys[((size_t)t*BN_ + bn)*D_ + tid] = hn;
}

// Qv = sum_a act * (ln(out) @ qval_w^T + qval_b)
__global__ __launch_bounds__(64) void k_qval(const float* __restrict__ ys, const float* __restrict__ actions,
    const float* __restrict__ qw, const float* __restrict__ qb, float* __restrict__ out_qv){
  __shared__ float o[D_]; __shared__ float red[64]; __shared__ float part[16];
  int row = blockIdx.x, tid = threadIdx.x;
  int bt = row / N_, n = row % N_;
  int b = bt / T_, t = bt % T_;
  int bn = b*N_ + n;
  float v   = ys[((size_t)t*BN_ + bn)*D_ + tid];
  float mu  = block_sum(v, red, 64) * (1.0f/64.0f);
  float d   = v - mu;
  float var = block_sum(d*d, red, 64) * (1.0f/64.0f);
  o[tid] = d * rsqrtf(var + 1e-5f);
  __syncthreads();
  if (tid < 16) part[tid] = 0.f;
  __syncthreads();
  if (tid < A_){
    float q = qb[tid];
    const float* wr = qw + (size_t)tid*D_;
    for (int k = 0; k < D_; ++k) q += o[k]*wr[k];
    part[tid] = q * actions[(size_t)row*A_ + tid];
  }
  __syncthreads();
  if (tid == 0){
    float s = 0.f;
    for (int a = 0; a < A_; ++a) s += part[a];
    out_qv[row] = s;
  }
}

__global__ void k_copy(const float* __restrict__ src, float* __restrict__ dst, int n){
  int i = blockIdx.x*blockDim.x + threadIdx.x;
  if (i < n) dst[i] = src[i];
}

// ================= launch =================
extern "C" void kernel_launch(void* const* d_in, const int* in_sizes, int n_in,
                              void* d_out, int out_size, void* d_ws, size_t ws_size,
                              hipStream_t stream){
  const float* states  = (const float*)d_in[0];
  const float* enemy   = (const float*)d_in[1];
  const float* actions = (const float*)d_in[2];
  const float* rnn_h0  = (const float*)d_in[3];
  const float* w_ally1 = (const float*)d_in[4];
  const float* b_ally1 = (const float*)d_in[5];
  const float* w_enemy = (const float*)d_in[6];
  const float* b_enemy = (const float*)d_in[7];
  const float* w_aact  = (const float*)d_in[8];
  const float* b_aact  = (const float*)d_in[9];
  const float* key_w   = (const float*)d_in[10];
  const float* key_b   = (const float*)d_in[11];
  const float* query_w = (const float*)d_in[12];
  const float* query_b = (const float*)d_in[13];
  const float* av_w    = (const float*)d_in[14];
  const float* av_b    = (const float*)d_in[15];
  const float* hard_w  = (const float*)d_in[16];
  const float* hard_b  = (const float*)d_in[17];
  const float* hlin_w  = (const float*)d_in[18];
  const float* hlin_b  = (const float*)d_in[19];
  const float* ffn1_w  = (const float*)d_in[20];
  const float* ffn1_b  = (const float*)d_in[21];
  const float* ffn2_w  = (const float*)d_in[22];
  const float* ffn2_b  = (const float*)d_in[23];
  const float* com1_w  = (const float*)d_in[24];
  const float* com1_b  = (const float*)d_in[25];
  const float* com2_w  = (const float*)d_in[26];
  const float* com2_b  = (const float*)d_in[27];
  const float* gru_wih = (const float*)d_in[28];
  const float* gru_whh = (const float*)d_in[29];
  const float* gru_bih = (const float*)d_in[30];
  const float* gru_bhh = (const float*)d_in[31];
  const float* qval_w  = (const float*)d_in[32];
  const float* qval_b  = (const float*)d_in[33];

  float* wsf = (float*)d_ws;
  float* st   = wsf + OFF_ST;
  float* se   = wsf + OFF_SE;
  float* oae  = wsf + OFF_OAE;
  float* Kp   = wsf + OFF_KP;
  float* Qp   = wsf + OFF_QP;
  float* Lq   = wsf + OFF_LQ;
  float* Lk   = wsf + OFF_LK;
  float* wsft = wsf + OFF_WSF;
  float* Vp   = wsf + OFF_VP;
  float* aggb = wsf + OFF_AGG;
  float* x1   = wsf + OFF_X1;
  float* x2   = wsf + OFF_X2;
  float* ee   = wsf + OFF_EE;
  float* gx   = wsf + OFF_GX;
  float* hst  = wsf + OFF_HST;
  float* ys   = wsf + OFF_YS;
  _Float16* w1h = (_Float16*)(wsf + OFF_F16);
  _Float16* w2h = w1h + FFNH*D_;

  float* out   = (float*)d_out;
  float* oQv   = out + OQV;
  float* oW    = out + OW;
  float* oScr  = out + OS;
  float* oHT   = out + OH;

  k_cvt   <<<512, 256, 0, stream>>>(ffn1_w, ffn2_w, w1h, w2h);
  k_st_se <<<NROW, 128, 0, stream>>>(states, w_ally1, b_ally1, st, se);
  k_oae   <<<NROW, 128, 0, stream>>>(st, actions, w_aact, b_aact, oae);
  k_ee    <<<BT_,  128, 0, stream>>>(enemy, w_enemy, b_enemy, ee);
  k_proj  <<<NROW, 64,  0, stream>>>(se, oae, key_w, key_b, query_w, query_b,
                                     av_w, av_b, hard_w, hlin_w, Kp, Qp, Vp, Lq, Lk);
  k_attn  <<<BT_*H_*N_, 32, 0, stream>>>(Kp, Qp, Vp, Lq, Lk, hard_b, hlin_w, hlin_b,
                                         wsft, aggb, oW, oScr);
  k_x1    <<<NROW, 64,  0, stream>>>(oae, aggb, x1);
  k_ffn   <<<NROW/16, 256, 0, stream>>>(x1, w1h, ffn1_b, w2h, ffn2_b, x2);
  k_common<<<NROW, 128, 0, stream>>>(se, ee, x2, com1_w, com1_b, com2_w, com2_b,
                                     gru_wih, gru_bih, gx);
  for (int t = 0; t < T_; ++t){
    const float* prev = (t == 0) ? rnn_h0 : (const float*)hst;
    k_gru <<<BN_, 64, 0, stream>>>(prev, gx, gru_whh, gru_bhh, hst, ys, t);
  }
  k_qval  <<<NROW, 64, 0, stream>>>(ys, actions, qval_w, qval_b, oQv);
  k_copy  <<<(BN_*D_ + 255)/256, 256, 0, stream>>>(hst, oHT, BN_*D_);
}
